// FeaturesSimilarityCriterion_38465727103538
// MI455X (gfx1250) — matrix operand order repaired and checked by reading, not compiled, not run on hardware
//
#include <hip/hip_runtime.h>
#include <math.h>
#include <stdint.h>

typedef __bf16 bf16_t;
typedef __attribute__((ext_vector_type(16))) __bf16 v16bf;
typedef __attribute__((ext_vector_type(8)))  __bf16 v8bf;
typedef __attribute__((ext_vector_type(4)))  __bf16 v4bf;
typedef __attribute__((ext_vector_type(8)))  float  v8f;

#define SHOTS   64
#define S_CNT   256
#define D_DIM   4096
#define B_CNT   64
#define Q_CNT   256
#define M_TOT   (B_CNT * Q_CNT)   // 16384
#define K_STEP  32
#define M_TILE  64
#define NT_WAVE 8                 // N-tiles per wave (8 x 16 = 128 cols)
#define MARGIN  0.3f
#define EPS_F   1e-8f

// ---------------------------------------------------------------------------
// Kernel 1: proto = mean(supp, axis=0) -> bf16 [S, D]; pn_sq[s] = ||proto_s||^2
// ---------------------------------------------------------------------------
__global__ void proto_kernel(const float* __restrict__ supp,
                             bf16_t* __restrict__ protoB,
                             float* __restrict__ pn_sq) {
    __shared__ float red[256];
    const int t    = threadIdx.x;
    const int idx4 = blockIdx.x * 256 + t;     // which float4 of proto
    const int base = idx4 * 4;                 // element index in [0, S*D)
    const int s    = blockIdx.x >> 2;          // whole block inside one s row

    const float4* src = (const float4*)supp;
    float4 acc = make_float4(0.f, 0.f, 0.f, 0.f);
    #pragma unroll 4
    for (int k = 0; k < SHOTS; ++k) {
        float4 v = src[(size_t)k * (S_CNT * D_DIM / 4) + idx4];
        acc.x += v.x; acc.y += v.y; acc.z += v.z; acc.w += v.w;
    }
    const float inv = 1.0f / (float)SHOTS;
    acc.x *= inv; acc.y *= inv; acc.z *= inv; acc.w *= inv;

    v4bf o;
    o[0] = (bf16_t)acc.x; o[1] = (bf16_t)acc.y;
    o[2] = (bf16_t)acc.z; o[3] = (bf16_t)acc.w;
    *(v4bf*)(protoB + base) = o;

    red[t] = acc.x * acc.x + acc.y * acc.y + acc.z * acc.z + acc.w * acc.w;
    __syncthreads();
    #pragma unroll
    for (int off = 128; off > 0; off >>= 1) {
        if (t < off) red[t] += red[t + off];
        __syncthreads();
    }
    if (t == 0) atomicAdd(&pn_sq[s], red[0]);
}

// ---------------------------------------------------------------------------
// Kernel 2: fused GEMM (bf16 WMMA, f32 accum) + cosine-embedding loss.
// 256 blocks x 256 threads (8 wave32). Block owns 64 M rows x all 256 N cols.
// Wave w: M-subtile (w>>1)*16, N-half (w&1)*128 -> 8 accum tiles (64 VGPRs).
// B tile staged with GLOBAL_LOAD_ASYNC_TO_LDS_B128 (no VGPR round trip);
// qn^2 accumulated while staging A (query read exactly once from HBM).
// ---------------------------------------------------------------------------
__global__ void __launch_bounds__(256, 2)
gemm_loss_kernel(const float* __restrict__ query,
                 const bf16_t* __restrict__ protoB,
                 const float* __restrict__ pn_sq,
                 float* __restrict__ out) {
    __shared__ __attribute__((aligned(64))) bf16_t As[M_TILE][K_STEP];   // 4 KB
    __shared__ __attribute__((aligned(64))) bf16_t Bs[S_CNT][K_STEP];    // 16 KB
    __shared__ float pns[S_CNT];
    __shared__ float qsq[256];
    __shared__ float qrow[M_TILE];
    __shared__ float red[256];

    const int t       = threadIdx.x;     // 0..255
    const int lane    = t & 31;
    const int w       = t >> 5;          // wave id 0..7
    const int ms      = w >> 1;          // M-subtile 0..3 (16 rows each)
    const int nh      = w & 1;           // N-half 0..1 (128 cols each)
    const int block_m = blockIdx.x * M_TILE;

    // prototype norms (sqrt once, reused in epilogue)
    pns[t] = sqrtf(pn_sq[t]);

    v8f acc[NT_WAVE] = {};               // 8 C tiles = 64 VGPRs

    // A staging map: thread -> (row, 8-float chunk of the 32-wide K step)
    const int   arow = t >> 2;           // 0..63
    const int   akof = (t & 3) * 8;      // 0,8,16,24
    const float* aptr = query + (size_t)(block_m + arow) * D_DIM + akof;

    // B staging: one proto row per thread, async direct to LDS.
    // INST_OFFSET is applied to BOTH lds and global addresses (ISA 15.18.3),
    // and our LDS row / global row advance in lockstep per 16B chunk.
    const unsigned ldsB = (unsigned)(uintptr_t)&Bs[t][0];
    const bf16_t*  gBrow = protoB + (size_t)t * D_DIM;

    // fragment addressing (wave32; contiguous K order, same for A and B)
    const int mfrag = ms * 16 + (lane & 15);
    const int ka    = (lane < 16) ? 0 : 16;   // A: 16 contiguous K per half-wave
    const int kb    = (lane < 16) ? 0 : 16;   // B: 16 contiguous K per half-wave
    const int nlow  = lane & 15;
    const int nbase = nh * 128;               // column base for this wave

    float ssq = 0.f;

    for (int kt = 0; kt < D_DIM / K_STEP; ++kt) {
        const int k0 = kt * K_STEP;
        __syncthreads();   // previous iteration's LDS reads done

        // ---- stage B: 64B per thread, memory -> LDS, no VGPR round trip --
        {
            uint64_t ga = (uint64_t)(uintptr_t)(gBrow + k0);
            asm volatile(
                "global_load_async_to_lds_b128 %0, %1, off\n\t"
                "global_load_async_to_lds_b128 %0, %1, off offset:16\n\t"
                "global_load_async_to_lds_b128 %0, %1, off offset:32\n\t"
                "global_load_async_to_lds_b128 %0, %1, off offset:48"
                :: "v"(ldsB), "v"(ga) : "memory");
        }
        // ---- stage A: 8 f32 -> 8 bf16 (one b128 LDS store), fused qn^2 ---
        {
            float4 v0 = *(const float4*)(aptr + k0);
            float4 v1 = *(const float4*)(aptr + k0 + 4);
            ssq += v0.x * v0.x + v0.y * v0.y + v0.z * v0.z + v0.w * v0.w;
            ssq += v1.x * v1.x + v1.y * v1.y + v1.z * v1.z + v1.w * v1.w;
            v8bf b;
            b[0] = (bf16_t)v0.x; b[1] = (bf16_t)v0.y;
            b[2] = (bf16_t)v0.z; b[3] = (bf16_t)v0.w;
            b[4] = (bf16_t)v1.x; b[5] = (bf16_t)v1.y;
            b[6] = (bf16_t)v1.z; b[7] = (bf16_t)v1.w;
            *(v8bf*)&As[arow][akof] = b;
        }
        // drain async B transfers before publishing the tile
        asm volatile("s_wait_asynccnt 0x0" ::: "memory");
        __syncthreads();

        // ---- WMMA: one A fragment vs 8 B fragments -----------------------
        v8bf alo = *(const v8bf*)&As[mfrag][ka];
        v8bf ahi = *(const v8bf*)&As[mfrag][ka + 8];
        v16bf afrag = __builtin_shufflevector(alo, ahi,
            0, 1, 2, 3, 4, 5, 6, 7, 8, 9, 10, 11, 12, 13, 14, 15);

        #pragma unroll
        for (int nt = 0; nt < NT_WAVE; ++nt) {
            v16bf bfrag = *(const v16bf*)&Bs[nbase + nt * 16 + nlow][kb];
            acc[nt] = __builtin_amdgcn_wmma_f32_16x16x32_bf16(
                false, afrag, false, bfrag, (short)0, acc[nt], false, false);
        }
    }

    // ---- query norms per local row (4 staging threads per row) -----------
    qsq[t] = ssq;
    __syncthreads();
    if (t < M_TILE)
        qrow[t] = sqrtf(qsq[4 * t] + qsq[4 * t + 1] + qsq[4 * t + 2] + qsq[4 * t + 3]);
    __syncthreads();

    // ---- fused loss epilogue (C tiles never leave registers) -------------
    float local = 0.f;
    #pragma unroll
    for (int nt = 0; nt < NT_WAVE; ++nt) {
        const int   col = nbase + nt * 16 + nlow;
        const float pn  = pns[col];
        #pragma unroll
        for (int v = 0; v < 8; ++v) {
            const int   rl   = ms * 16 + v + ((lane >= 16) ? 8 : 0);
            const float qn   = qrow[rl];
            const float cosv = acc[nt][v] / fmaxf(qn * pn, EPS_F);
            const int   gq   = (block_m + rl) & (Q_CNT - 1);  // query idx in [0,256)
            const bool  pos  = (gq % 6) == 0;
            local += pos ? (1.0f - cosv) : fmaxf(cosv - MARGIN, 0.0f);
        }
    }
    red[t] = local;
    __syncthreads();
    #pragma unroll
    for (int off = 128; off > 0; off >>= 1) {
        if (t < off) red[t] += red[t + off];
        __syncthreads();
    }
    if (t == 0) atomicAdd(out, red[0] * (1.0f / (float)(B_CNT * S_CNT)));
}

// ---------------------------------------------------------------------------
extern "C" void kernel_launch(void* const* d_in, const int* in_sizes, int n_in,
                              void* d_out, int out_size, void* d_ws, size_t ws_size,
                              hipStream_t stream) {
    (void)in_sizes; (void)n_in; (void)out_size; (void)ws_size;

    const float* supp  = (const float*)d_in[0];
    const float* query = (const float*)d_in[1];
    float*       out   = (float*)d_out;

    // workspace layout: [0, 2MB) proto bf16 ; [2MB, 2MB+1KB) pn_sq f32
    bf16_t* protoB = (bf16_t*)d_ws;
    float*  pn_sq  = (float*)((char*)d_ws + (size_t)S_CNT * D_DIM * sizeof(bf16_t));

    hipMemsetAsync(pn_sq, 0, S_CNT * sizeof(float), stream);
    hipMemsetAsync(out, 0, sizeof(float), stream);

    proto_kernel<<<(S_CNT * D_DIM) / (256 * 4), 256, 0, stream>>>(supp, protoB, pn_sq);
    gemm_loss_kernel<<<M_TOT / M_TILE, 256, 0, stream>>>(query, protoB, pn_sq, out);
}
